// PiCANet_G_29042568855923
// MI455X (gfx1250) — matrix operand order repaired and checked by reading, not compiled, not run on hardware
//
#include <hip/hip_runtime.h>
#include <math.h>

// ---------------- types ----------------
typedef __bf16 bf16_t;
typedef __attribute__((ext_vector_type(16))) __bf16 v16bf;
typedef __attribute__((ext_vector_type(8)))  float  v8f;
typedef __attribute__((ext_vector_type(4)))  unsigned int uint4v;

#define BB   64
#define CIN  512
#define SZ   28
#define HID  256
#define NSEQ (BB * SZ)        // 1792 sequences per LSTM sweep
#define NTOK (BB * SZ * SZ)   // 50176 pixels

// =====================================================================
// Generic bf16 WMMA GEMM:  C[m,n] = sum_k A[m,k] * W[n,k]  (+ bias[n])
// A: bf16 [M x K] row-major (lda), W: bf16 [N x K] row-major (ldw),
// C: f32 [M x ldc]. M must be a multiple of 64, K a multiple of 32.
// N is guarded. Batched via blockIdx.z with element strides sA/sW/sC.
// Block: 256 threads = 8 wave32s (2x4 grid), tile 64(M) x 256(N),
// K-step 32. Each wave computes a 32x64 tile = 2x4 WMMA fragments
// (8 x v_wmma_f32_16x16x32_bf16 per K-step, 2 A-frags + 4 B-frags).
// =====================================================================
__global__ __launch_bounds__(256) void k_wmma_gemm(
    const bf16_t* __restrict__ A, const bf16_t* __restrict__ W,
    float* __restrict__ C, const float* __restrict__ bias,
    int N, int K, int lda, int ldw, int ldc,
    long long sA, long long sW, long long sC)
{
  __shared__ __align__(16) bf16_t As[64 * 32];    //  4 KB
  __shared__ __align__(16) bf16_t Bs[256 * 32];   // 16 KB

  const int tid   = threadIdx.x;
  const int m_blk = blockIdx.y * 64;
  const int n_blk = blockIdx.x * 256;
  A += (long long)blockIdx.z * sA;
  W += (long long)blockIdx.z * sW;
  C += (long long)blockIdx.z * sC;

  const int wave  = tid >> 5, lane = tid & 31;
  const int wm    = wave >> 2, wn = wave & 3;    // 2x4 wave grid
  const int lmod  = lane & 15, lhalf = lane >> 4;

  v8f acc[2][4] = {};

  for (int k0 = 0; k0 < K; k0 += 32) {
    // ---- stage A tile 64x32 (one b128 per thread) ----
    {
      int row = tid >> 2, cg = (tid & 3) << 3;
      *(uint4v*)&As[row * 32 + cg] =
          *(const uint4v*)(A + (long long)(m_blk + row) * lda + k0 + cg);
    }
    // ---- stage B tile 256x32 (four b128 per thread, N-guarded) ----
#pragma unroll
    for (int i = 0; i < 4; ++i) {
      int idx = tid + i * 256;
      int n = idx >> 2, cg = (idx & 3) << 3;
      uint4v val = {0u, 0u, 0u, 0u};
      int gn = n_blk + n;
      if (gn < N)
        val = *(const uint4v*)(W + (long long)gn * ldw + k0 + cg);
      *(uint4v*)&Bs[n * 32 + cg] = val;
    }
    __syncthreads();

    // ---- gather fragments per gfx1250 16-bit WMMA lane layout ----
    union Frag { unsigned u[8]; v16bf v; };
    Frag a[2], b[4];
#pragma unroll
    for (int ti = 0; ti < 2; ++ti) {
      const unsigned* p = (const unsigned*)&As[(wm * 32 + ti * 16 + lmod) * 32];
#pragma unroll
      for (int j = 0; j < 8; ++j)      // A: lanes<16 K={0..7,16..23}, lanes>=16 +8
        a[ti].u[j] = p[(j < 4 ? j : j + 4) + lhalf * 4];
    }
#pragma unroll
    for (int tj = 0; tj < 4; ++tj) {
      const unsigned* p = (const unsigned*)&Bs[(wn * 64 + tj * 16 + lmod) * 32];
#pragma unroll
      for (int j = 0; j < 8; ++j)      // B: lanes<16 K={0..7,16..23}, lanes>=16 +8
        b[tj].u[j] = p[(j < 4 ? j : j + 4) + lhalf * 4];
    }

#pragma unroll
    for (int ti = 0; ti < 2; ++ti)
#pragma unroll
      for (int tj = 0; tj < 4; ++tj)
        acc[ti][tj] = __builtin_amdgcn_wmma_f32_16x16x32_bf16(
            false, a[ti].v, false, b[tj].v, (short)0, acc[ti][tj],
            false, false);
    __syncthreads();
  }

  // ---- epilogue: C layout lane L, VGPR r -> M=r+8*(L/16), N=L%16 ----
#pragma unroll
  for (int ti = 0; ti < 2; ++ti) {
#pragma unroll
    for (int tj = 0; tj < 4; ++tj) {
      int col = n_blk + wn * 64 + tj * 16 + lmod;
      if (col < N) {
#pragma unroll
        for (int r = 0; r < 8; ++r) {
          int row = m_blk + wm * 32 + ti * 16 + r + 8 * lhalf;
          float v = acc[ti][tj][r];
          if (bias) v += bias[col];
          C[(long long)row * ldc + col] = v;
        }
      }
    }
  }
}

// =====================================================================
// LSTM gate fusion: g = P[r, t] + G[r] + bih + bhh; torch gate order i,f,g,o
// Writes c (f32), h (bf16 state for next step's GEMM), and the feature
// output at a generic strided location (fp32 and/or bf16).
// =====================================================================
__device__ __forceinline__ float sigm(float x) { return 1.0f / (1.0f + __expf(-x)); }

__global__ __launch_bounds__(256) void k_lstm_gates(
    const float* __restrict__ P, const float* __restrict__ G,
    const float* __restrict__ bih, const float* __restrict__ bhh,
    float* __restrict__ cst, bf16_t* __restrict__ hst,
    float* __restrict__ outF32, bf16_t* __restrict__ outBf16,
    int t, int dir, int s_hi, int s_lo, int s_t)
{
  int idx = blockIdx.x * 256 + threadIdx.x;
  if (idx >= NSEQ * HID) return;
  int r = idx >> 8, j = idx & 255;

  const float* Pr = P + ((long long)r * SZ + t) * 1024;
  const float* Gr = G + (long long)r * 1024;

  float gi = Pr[j]       + Gr[j]       + bih[j]       + bhh[j];
  float gf = Pr[256 + j] + Gr[256 + j] + bih[256 + j] + bhh[256 + j];
  float gg = Pr[512 + j] + Gr[512 + j] + bih[512 + j] + bhh[512 + j];
  float go = Pr[768 + j] + Gr[768 + j] + bih[768 + j] + bhh[768 + j];

  float c = sigm(gf) * cst[idx] + sigm(gi) * tanhf(gg);
  float h = sigm(go) * tanhf(c);
  cst[idx] = c;
  hst[idx] = (bf16_t)h;

  long long o = (long long)(r / SZ) * s_hi + (long long)(r % SZ) * s_lo +
                (long long)t * s_t + dir * HID + j;
  if (outF32)  outF32[o]  = h;
  if (outBf16) outBf16[o] = (bf16_t)h;
}

// ---------------- elementwise helpers ----------------
__global__ void k_f32_to_bf16(const float* __restrict__ in, bf16_t* __restrict__ out, int n) {
  int i = blockIdx.x * 256 + threadIdx.x;
  if (i < n) out[i] = (bf16_t)in[i];
}

// x[B,C,H,W] f32 -> xv[(b*28+h)*28+w, c] bf16
__global__ void k_perm_xv(const float* __restrict__ x, bf16_t* __restrict__ o) {
  int i = blockIdx.x * 256 + threadIdx.x;
  if (i >= NTOK * CIN) return;
  int c = i & 511; int rest = i >> 9;
  int w = rest % SZ; rest /= SZ; int h = rest % SZ; int b = rest / SZ;
  o[i] = (bf16_t)x[(((b * CIN + c) * SZ + h) * SZ) + w];
}

// v[B,H,W,512] f32 -> hin[(b*28+w)*28+h, c] bf16
__global__ void k_perm_hin(const float* __restrict__ v, bf16_t* __restrict__ o) {
  int i = blockIdx.x * 256 + threadIdx.x;
  if (i >= NTOK * CIN) return;
  int c = i & 511; int rest = i >> 9;
  int h = rest % SZ; rest /= SZ; int w = rest % SZ; int b = rest / SZ;
  o[i] = (bf16_t)v[(((b * SZ + h) * SZ + w) * CIN) + c];
}

// fc_W [512,100] f32 -> fcWt [100,512] bf16 (transposed to [N,K])
__global__ void k_fcwt(const float* __restrict__ fcW, bf16_t* __restrict__ o) {
  int i = blockIdx.x * 256 + threadIdx.x;
  if (i >= 100 * 512) return;
  int n = i / 512, k = i % 512;
  o[i] = (bf16_t)fcW[k * 100 + n];
}

// patch[b,c,k] = x[b,c,3*(k/10),3*(k%10)], k padded 100 -> 128 with zeros
__global__ void k_patch(const float* __restrict__ x, bf16_t* __restrict__ o) {
  int i = blockIdx.x * 256 + threadIdx.x;
  if (i >= BB * CIN * 128) return;
  int k = i & 127; int rest = i >> 7; int c = rest & 511; int b = rest >> 9;
  float v = 0.f;
  if (k < 100)
    v = x[(((b * CIN + c) * SZ + (k / 10) * 3) * SZ) + (k % 10) * 3];
  o[i] = (bf16_t)v;
}

__global__ void k_zero_state(float* __restrict__ c, bf16_t* __restrict__ h) {
  int i = blockIdx.x * 256 + threadIdx.x;
  if (i >= NSEQ * HID) return;
  c[i] = 0.f; h[i] = (bf16_t)0.f;
}

// wave-per-row softmax over 100 taps -> bf16 kern [row,128] (tail zeroed)
__global__ __launch_bounds__(256) void k_softmax(const float* __restrict__ logits,
                                                 bf16_t* __restrict__ kern) {
  int wave = threadIdx.x >> 5, lane = threadIdx.x & 31;
  int row = blockIdx.x * 8 + wave;               // NTOK % 8 == 0
  const float* L = logits + (long long)row * 100;
  float v[4]; float mx = -1e30f;
#pragma unroll
  for (int i = 0; i < 4; ++i) {
    int c = lane + i * 32;
    v[i] = (c < 100) ? L[c] : -1e30f;
    mx = fmaxf(mx, v[i]);
  }
  for (int m = 16; m > 0; m >>= 1) mx = fmaxf(mx, __shfl_xor(mx, m));
  float sum = 0.f;
#pragma unroll
  for (int i = 0; i < 4; ++i) {
    v[i] = (lane + i * 32 < 100) ? __expf(v[i] - mx) : 0.f;
    sum += v[i];
  }
  for (int m = 16; m > 0; m >>= 1) sum += __shfl_xor(sum, m);
  float inv = 1.0f / sum;
  bf16_t* O = kern + (long long)row * 128;
#pragma unroll
  for (int i = 0; i < 4; ++i) {
    int c = lane + i * 32;
    if (c < 100) O[c] = (bf16_t)(v[i] * inv);
  }
  if (lane < 28) O[100 + lane] = (bf16_t)0.f;
}

// ---------------- host side ----------------
static inline void launch_gemm(hipStream_t s, const bf16_t* A, const bf16_t* W,
                               float* C, const float* bias, int M, int N, int K,
                               int lda, int ldw, int ldc,
                               long long sA, long long sW, long long sC, int batch) {
  dim3 g((N + 255) / 256, M / 64, batch);
  k_wmma_gemm<<<g, dim3(256), 0, s>>>(A, W, C, bias, N, K, lda, ldw, ldc, sA, sW, sC);
}

extern "C" void kernel_launch(void* const* d_in, const int* in_sizes, int n_in,
                              void* d_out, int out_size, void* d_ws, size_t ws_size,
                              hipStream_t stream) {
  const float* x = (const float*)d_in[0];
  // dict order: x, then (Wih,Whh,bih,bhh) for vf, vb, hf, hb, then fc_W, fc_b
  const float* Wih_f32[4] = {(const float*)d_in[1], (const float*)d_in[5],
                             (const float*)d_in[9], (const float*)d_in[13]};
  const float* Whh_f32[4] = {(const float*)d_in[2], (const float*)d_in[6],
                             (const float*)d_in[10], (const float*)d_in[14]};
  const float* bih[4] = {(const float*)d_in[3], (const float*)d_in[7],
                         (const float*)d_in[11], (const float*)d_in[15]};
  const float* bhh[4] = {(const float*)d_in[4], (const float*)d_in[8],
                         (const float*)d_in[12], (const float*)d_in[16]};
  const float* fcW = (const float*)d_in[17];
  const float* fcB = (const float*)d_in[18];
  float* out = (float*)d_out;

  // ---- workspace carve-out (256B aligned cursor) ----
  char* ws = (char*)d_ws;
  size_t off = 0;
  auto alloc = [&](size_t bytes) -> void* {
    void* p = ws + off;
    off = (off + bytes + 255) & ~(size_t)255;
    return p;
  };
  bf16_t* xv     = (bf16_t*)alloc((size_t)NTOK * CIN * 2);   // vertical seq, bf16
  bf16_t* hin    = (bf16_t*)alloc((size_t)NTOK * CIN * 2);   // horizontal seq, bf16
  bf16_t* feat   = (bf16_t*)alloc((size_t)NTOK * CIN * 2);   // final feature, bf16
  float*  vbuf   = (float*) alloc((size_t)NTOK * CIN * 4);   // vertical feature, f32
  float*  P      = (float*) alloc((size_t)NTOK * 1024 * 4);  // hoisted Wih pre-acts
  float*  G      = (float*) alloc((size_t)NSEQ * 1024 * 4);  // per-step Whh GEMM
  bf16_t* hst    = (bf16_t*)alloc((size_t)NSEQ * HID * 2);
  float*  cst    = (float*) alloc((size_t)NSEQ * HID * 4);
  float*  logits = (float*) alloc((size_t)NTOK * 100 * 4);
  bf16_t* kern   = (bf16_t*)alloc((size_t)BB * 784 * 128 * 2);
  bf16_t* patch  = (bf16_t*)alloc((size_t)BB * CIN * 128 * 2);
  bf16_t* Wih_b[4], *Whh_b[4];
  for (int i = 0; i < 4; ++i) {
    Wih_b[i] = (bf16_t*)alloc((size_t)1024 * 512 * 2);
    Whh_b[i] = (bf16_t*)alloc((size_t)1024 * 256 * 2);
  }
  bf16_t* fcWt = (bf16_t*)alloc((size_t)100 * 512 * 2);

  auto blk = [](long long n) { return (unsigned)((n + 255) / 256); };

  // ---- weight conversion + input permute + patch extraction ----
  for (int i = 0; i < 4; ++i) {
    k_f32_to_bf16<<<blk(1024 * 512), 256, 0, stream>>>(Wih_f32[i], Wih_b[i], 1024 * 512);
    k_f32_to_bf16<<<blk(1024 * 256), 256, 0, stream>>>(Whh_f32[i], Whh_b[i], 1024 * 256);
  }
  k_fcwt<<<blk(100 * 512), 256, 0, stream>>>(fcW, fcWt);
  k_perm_xv<<<blk((long long)NTOK * CIN), 256, 0, stream>>>(x, xv);
  k_patch<<<blk((long long)BB * CIN * 128), 256, 0, stream>>>(x, patch);

  // ---- vertical bi-LSTM (seq along W, feature -> vbuf f32) ----
  for (int dir = 0; dir < 2; ++dir) {
    launch_gemm(stream, xv, Wih_b[dir], P, nullptr,
                NTOK, 1024, 512, 512, 512, 1024, 0, 0, 0, 1);
    k_zero_state<<<blk((long long)NSEQ * HID), 256, 0, stream>>>(cst, hst);
    for (int s = 0; s < SZ; ++s) {
      launch_gemm(stream, hst, Whh_b[dir], G, nullptr,
                  NSEQ, 1024, 256, 256, 256, 1024, 0, 0, 0, 1);
      int t = dir ? (SZ - 1 - s) : s;
      k_lstm_gates<<<blk((long long)NSEQ * HID), 256, 0, stream>>>(
          P, G, bih[dir], bhh[dir], cst, hst, vbuf, (bf16_t*)nullptr,
          t, dir, SZ * SZ * 512, SZ * 512, 512);
    }
  }

  // ---- horizontal bi-LSTM (seq along H, feature -> feat bf16) ----
  k_perm_hin<<<blk((long long)NTOK * CIN), 256, 0, stream>>>(vbuf, hin);
  for (int dir = 0; dir < 2; ++dir) {
    int wi = 2 + dir;
    launch_gemm(stream, hin, Wih_b[wi], P, nullptr,
                NTOK, 1024, 512, 512, 512, 1024, 0, 0, 0, 1);
    k_zero_state<<<blk((long long)NSEQ * HID), 256, 0, stream>>>(cst, hst);
    for (int s = 0; s < SZ; ++s) {
      launch_gemm(stream, hst, Whh_b[wi], G, nullptr,
                  NSEQ, 1024, 256, 256, 256, 1024, 0, 0, 0, 1);
      int t = dir ? (SZ - 1 - s) : s;
      k_lstm_gates<<<blk((long long)NSEQ * HID), 256, 0, stream>>>(
          P, G, bih[wi], bhh[wi], cst, hst, (float*)nullptr, feat,
          t, dir, SZ * SZ * 512, 512, SZ * 512);
    }
  }

  // ---- FC (512 -> 100) + softmax -> bf16 attention kernel ----
  launch_gemm(stream, feat, fcWt, logits, fcB,
              NTOK, 100, 512, 512, 512, 100, 0, 0, 0, 1);
  k_softmax<<<NTOK / 8, 256, 0, stream>>>(logits, kern);

  // ---- final batched contraction: out[b] = patch[b] (512x128) x kern[b]^T ----
  launch_gemm(stream, patch, kern, out, nullptr,
              512, 784, 128, 128, 128, 784,
              (long long)512 * 128, (long long)784 * 128, (long long)512 * 784, BB);
}
